// GraphConvNet_54116587930156
// MI455X (gfx1250) — matrix operand order repaired and checked by reading, not compiled, not run on hardware
//
#include <hip/hip_runtime.h>
#include <stdint.h>

// ---------------------------------------------------------------------------
// GraphNet (Encode-Process-Decode) for MI455X / gfx1250.
//   - all MLP GEMMs via v_wmma_f32_16x16x32_f16 (f16 in, f32 accumulate)
//   - M=32 per wave: two 16-row tiles share every B fragment (2 WMMAs per
//     global_load_b128 pair; weight stream lives in L2, ~300 KB total)
//   - no occupancy cap: LDS (53 KB/block) already limits the WGP to
//     ~3 waves/SIMD, so the ~300-VGPR allocation is spill-free and right
//   - edge features carried between steps as f16 (halves HBM traffic)
//   - segment_sum via global_atomic_add_f32
// Input order assumed = setup_inputs() dict insertion order, params flattened
// recursively in insertion order:
//   0 nodes(N*3) 1 globals(8) 2 senders(E) 3 receivers(E)
//   4 encW(3x128) 5 encB 6 decW(128x3) 7 decB
//   per step s (base = 8 + 14*s):
//     +0..5  edge MLP  (W1,b1,W2,b2,W3,b3)   W1: (264|392)x128
//     +6..11 node MLP  (W1,b1,W2,b2,W3,b3)   W1: 264x128
//     +12 ln_scale  +13 ln_bias
// ---------------------------------------------------------------------------

#define L 128
#define G 8
#define WAVES 2          // waves per block in the WMMA MLP kernels

typedef _Float16 v16h __attribute__((ext_vector_type(16)));
typedef float    v8f  __attribute__((ext_vector_type(8)));

union AFrag { v16h v; uint32_t u[8]; };
union BFrag { v16h v; uint4 q[2]; };
union H4    { _Float16 h[4]; uint2 u; };

__device__ __forceinline__ float gelu_tanh(float x) {
  float u = 0.7978845608028654f * (x + 0.044715f * x * x * x);
  u = fminf(fmaxf(u, -10.0f), 10.0f);
  float e = __expf(2.0f * u);                     // v_exp_f32
  float r = __builtin_amdgcn_rcpf(e + 1.0f);      // v_rcp_f32 (no div expansion)
  float t = fmaf(-2.0f, r, 1.0f);                 // tanh(u)
  return 0.5f * x * (1.0f + t);
}

// One 32xKP @ KPx128 layer: A (two 16-row tiles) staged in LDS (row-major
// f16, stride KP), B prepacked in WMMA B-fragment order and streamed with a
// linearly advancing pointer. Bias folded into the C init.
__device__ __forceinline__ void wmma_layer_dual(const _Float16* __restrict__ Xs, int KP,
                                                int kchunks, const uint4* __restrict__ Wp,
                                                const float* __restrict__ bias,
                                                v8f acc0[8], v8f acc1[8], int lane) {
  const int m  = lane & 15;          // C column (and A row) for this lane
  const int kh = (lane >> 4) & 1;
#pragma unroll
  for (int nt = 0; nt < 8; ++nt) {
    float bv = bias[nt * 16 + m];
    v8f t = {bv, bv, bv, bv, bv, bv, bv, bv};
    acc0[nt] = t;
    acc1[nt] = t;
  }
  const uint4* wp = Wp + lane * 2;   // fragments are contiguous: 64 uint4 apart
  for (int kc = 0; kc < kchunks; ++kc) {
    AFrag a0, a1;
#pragma unroll
    for (int v = 0; v < 8; ++v) {
      // A k order used here (contiguous, same as B pack): lanes 0-15 K base 0, lanes 16-31 K base 16; VGPR v -> K + 2v
      int K = kc * 32 + kh * 16 + v * 2;
      a0.u[v] = *reinterpret_cast<const uint32_t*>(Xs + m * KP + K);
      a1.u[v] = *reinterpret_cast<const uint32_t*>(Xs + (m + 16) * KP + K);
    }
#pragma unroll
    for (int nt = 0; nt < 8; ++nt) {
      BFrag b;
      b.q[0] = wp[0];
      b.q[1] = wp[1];
      wp += 64;
      acc0[nt] = __builtin_amdgcn_wmma_f32_16x16x32_f16(
          false, a0.v, false, b.v, (short)0, acc0[nt], false, false);
      acc1[nt] = __builtin_amdgcn_wmma_f32_16x16x32_f16(
          false, a1.v, false, b.v, (short)0, acc1[nt], false, false);
    }
  }
}

// GELU + downconvert, write 32x128 f16 activation back into the LDS strip.
__device__ __forceinline__ void gelu_store_dual(_Float16* __restrict__ Xs, int KP,
                                                v8f acc0[8], v8f acc1[8], int lane) {
  const int n0 = lane & 15;
  const int kh = (lane >> 4) & 1;
#pragma unroll
  for (int nt = 0; nt < 8; ++nt) {
#pragma unroll
    for (int i = 0; i < 8; ++i) {
      float x0 = gelu_tanh(acc0[nt][i]);
      float x1 = gelu_tanh(acc1[nt][i]);
      Xs[(i + 8 * kh) * KP + nt * 16 + n0]      = (_Float16)x0;
      Xs[(16 + i + 8 * kh) * KP + nt * 16 + n0] = (_Float16)x1;
    }
  }
}

template <int KP>
__device__ __forceinline__ void mlp3_dual(_Float16* Xs,
                                          const uint4* W1, const float* b1,
                                          const uint4* W2, const float* b2,
                                          const uint4* W3, const float* b3,
                                          v8f out0[8], v8f out1[8], int lane) {
  v8f acc0[8], acc1[8];
  wmma_layer_dual(Xs, KP, KP / 32, W1, b1, acc0, acc1, lane);
  gelu_store_dual(Xs, KP, acc0, acc1, lane);
  wmma_layer_dual(Xs, KP, 4, W2, b2, acc0, acc1, lane);
  gelu_store_dual(Xs, KP, acc0, acc1, lane);
  wmma_layer_dual(Xs, KP, 4, W3, b3, out0, out1, lane);
}

// Epilogue for one 16-row half-tile of the edge MLP: scatter new_edges (f16)
// and atomically accumulate into recv_agg. Resolved at compile time per call
// site (no dynamic array select -> keeps outputs in registers).
__device__ __forceinline__ void edge_epilogue(const v8f (&oo)[8], int base16,
                                              const int* __restrict__ rcv,
                                              _Float16* __restrict__ edges,
                                              float* __restrict__ recv_agg,
                                              int E, int lane) {
  const int n0 = lane & 15;
  const int kh = (lane >> 4) & 1;
#pragma unroll
  for (int i = 0; i < 8; ++i) {
    const int e = base16 + i + 8 * kh;
    if (e < E) {
      const size_t erow = (size_t)e * L;
      const size_t rrow = (size_t)rcv[e] * L;
#pragma unroll
      for (int nt = 0; nt < 8; ++nt) {
        const float x = oo[nt][i];
        edges[erow + nt * 16 + n0] = (_Float16)x;
        atomicAdd(recv_agg + rrow + nt * 16 + n0, x);
      }
    }
  }
}

// Epilogue for one 16-row half-tile of the node MLP: skip-add and store fp32.
__device__ __forceinline__ void node_epilogue(const v8f (&oo)[8], int base16,
                                              const float* __restrict__ h,
                                              float* __restrict__ tmp,
                                              int N, int lane) {
  const int n0 = lane & 15;
  const int kh = (lane >> 4) & 1;
#pragma unroll
  for (int i = 0; i < 8; ++i) {
    const int n = base16 + i + 8 * kh;
    if (n < N) {
      const size_t nrow = (size_t)n * L;
#pragma unroll
      for (int nt = 0; nt < 8; ++nt) {
        tmp[nrow + nt * 16 + n0] = oo[nt][i] + h[nrow + nt * 16 + n0];
      }
    }
  }
}

// ---------------- edge MLP + scatter-add into recv_agg -------------------
template <int KP, bool HAS_EDGES>
__launch_bounds__(WAVES * 32)
__global__ void edge_mlp_kernel(const float* __restrict__ h, const float* __restrict__ g,
                                const int* __restrict__ snd, const int* __restrict__ rcv,
                                _Float16* __restrict__ edges, float* __restrict__ recv_agg,
                                const uint4* W1, const float* b1,
                                const uint4* W2, const float* b2,
                                const uint4* W3, const float* b3,
                                int E, int numTiles) {
  __shared__ _Float16 XsAll[WAVES][32 * KP];
  const int lane = threadIdx.x & 31;
  const int wid  = threadIdx.x >> 5;
  _Float16* Xs = XsAll[wid];
  const int nw = gridDim.x * WAVES;
  const int goff = KP - 32;  // globals at [goff, goff+8), zero-pad to KP
  for (int t = blockIdx.x * WAVES + wid; t < numTiles; t += nw) {
    const int base = t * 32;
    for (int r = 0; r < 32; ++r) {
      int e = base + r;
      if (e >= E) e = E - 1;          // clamp; tail rows masked in epilogue
      _Float16* row = Xs + r * KP;
      int off = 0;
      if (HAS_EDGES) {
        *reinterpret_cast<uint2*>(row + lane * 4) =
            *reinterpret_cast<const uint2*>(edges + (size_t)e * L + lane * 4);
        off = L;
      }
      {
        const float4 a = *reinterpret_cast<const float4*>(h + (size_t)snd[e] * L + lane * 4);
        H4 p; p.h[0] = (_Float16)a.x; p.h[1] = (_Float16)a.y;
              p.h[2] = (_Float16)a.z; p.h[3] = (_Float16)a.w;
        *reinterpret_cast<uint2*>(row + off + lane * 4) = p.u;
      }
      {
        const float4 a = *reinterpret_cast<const float4*>(h + (size_t)rcv[e] * L + lane * 4);
        H4 p; p.h[0] = (_Float16)a.x; p.h[1] = (_Float16)a.y;
              p.h[2] = (_Float16)a.z; p.h[3] = (_Float16)a.w;
        *reinterpret_cast<uint2*>(row + off + L + lane * 4) = p.u;
      }
      row[goff + lane] = (lane < G) ? (_Float16)g[lane] : (_Float16)0.0f;
    }
    v8f out0[8], out1[8];
    mlp3_dual<KP>(Xs, W1, b1, W2, b2, W3, b3, out0, out1, lane);
    edge_epilogue(out0, base,      rcv, edges, recv_agg, E, lane);
    edge_epilogue(out1, base + 16, rcv, edges, recv_agg, E, lane);
  }
}

// ---------------- node MLP + skip connection ------------------------------
__launch_bounds__(WAVES * 32)
__global__ void node_mlp_kernel(const float* __restrict__ h, const float* __restrict__ agg,
                                const float* __restrict__ g, float* __restrict__ tmp,
                                const uint4* W1, const float* b1,
                                const uint4* W2, const float* b2,
                                const uint4* W3, const float* b3,
                                int N, int numTiles) {
  constexpr int KP = 288;
  __shared__ _Float16 XsAll[WAVES][32 * KP];
  const int lane = threadIdx.x & 31;
  const int wid  = threadIdx.x >> 5;
  _Float16* Xs = XsAll[wid];
  const int nw = gridDim.x * WAVES;
  const int goff = KP - 32;
  for (int t = blockIdx.x * WAVES + wid; t < numTiles; t += nw) {
    const int base = t * 32;
    for (int r = 0; r < 32; ++r) {
      int n = base + r;
      if (n >= N) n = N - 1;
      _Float16* row = Xs + r * KP;
      {
        const float4 a = *reinterpret_cast<const float4*>(h + (size_t)n * L + lane * 4);
        H4 p; p.h[0] = (_Float16)a.x; p.h[1] = (_Float16)a.y;
              p.h[2] = (_Float16)a.z; p.h[3] = (_Float16)a.w;
        *reinterpret_cast<uint2*>(row + lane * 4) = p.u;
      }
      {
        const float4 a = *reinterpret_cast<const float4*>(agg + (size_t)n * L + lane * 4);
        H4 p; p.h[0] = (_Float16)a.x; p.h[1] = (_Float16)a.y;
              p.h[2] = (_Float16)a.z; p.h[3] = (_Float16)a.w;
        *reinterpret_cast<uint2*>(row + L + lane * 4) = p.u;
      }
      row[goff + lane] = (lane < G) ? (_Float16)g[lane] : (_Float16)0.0f;
    }
    v8f out0[8], out1[8];
    mlp3_dual<KP>(Xs, W1, b1, W2, b2, W3, b3, out0, out1, lane);
    node_epilogue(out0, base,      h, tmp, N, lane);
    node_epilogue(out1, base + 16, h, tmp, N, lane);
  }
}

// ---------------- layernorm (wave per row, wave32 shuffles) ---------------
__global__ void layernorm_kernel(const float* __restrict__ x, float* __restrict__ h,
                                 const float* __restrict__ scale,
                                 const float* __restrict__ bias, int N) {
  const int lane = threadIdx.x & 31;
  const int wid  = threadIdx.x >> 5;
  int row = blockIdx.x * (blockDim.x >> 5) + wid;
  const int stride = gridDim.x * (blockDim.x >> 5);
  for (; row < N; row += stride) {
    const float4 v = *reinterpret_cast<const float4*>(x + (size_t)row * L + lane * 4);
    float s = v.x + v.y + v.z + v.w;
    for (int m = 16; m >= 1; m >>= 1) s += __shfl_xor(s, m, 32);
    const float mean = s * (1.0f / 128.0f);
    const float dx = v.x - mean, dy = v.y - mean, dz = v.z - mean, dw = v.w - mean;
    float s2 = dx * dx + dy * dy + dz * dz + dw * dw;
    for (int m = 16; m >= 1; m >>= 1) s2 += __shfl_xor(s2, m, 32);
    const float inv = rsqrtf(s2 * (1.0f / 128.0f) + 1e-6f);
    const float4 sc = *reinterpret_cast<const float4*>(scale + lane * 4);
    const float4 bi = *reinterpret_cast<const float4*>(bias + lane * 4);
    float4 o;
    o.x = dx * inv * sc.x + bi.x;
    o.y = dy * inv * sc.y + bi.y;
    o.z = dz * inv * sc.z + bi.z;
    o.w = dw * inv * sc.w + bi.w;
    *reinterpret_cast<float4*>(h + (size_t)row * L + lane * 4) = o;
  }
}

// ---------------- encoder / decoder / fill / weight-pack ------------------
__global__ void encode_kernel(const float* __restrict__ nodes, const float* __restrict__ W,
                              const float* __restrict__ b, float* __restrict__ h, int N) {
  const int idx = blockIdx.x * blockDim.x + threadIdx.x;
  if (idx >= N * L) return;
  const int n = idx >> 7, c = idx & 127;
  h[idx] = fmaf(nodes[n * 3 + 0], W[c],
           fmaf(nodes[n * 3 + 1], W[L + c],
           fmaf(nodes[n * 3 + 2], W[2 * L + c], b[c])));
}

__global__ void decode_kernel(const float* __restrict__ h, const float* __restrict__ W,
                              const float* __restrict__ b, float* __restrict__ out, int N) {
  const int n = blockIdx.x * blockDim.x + threadIdx.x;
  if (n >= N) return;
  float a0 = b[0], a1 = b[1], a2 = b[2];
  for (int k = 0; k < L; k += 4) {
    const float4 v = *reinterpret_cast<const float4*>(h + (size_t)n * L + k);
    a0 = fmaf(v.x, W[(k + 0) * 3 + 0], a0); a1 = fmaf(v.x, W[(k + 0) * 3 + 1], a1); a2 = fmaf(v.x, W[(k + 0) * 3 + 2], a2);
    a0 = fmaf(v.y, W[(k + 1) * 3 + 0], a0); a1 = fmaf(v.y, W[(k + 1) * 3 + 1], a1); a2 = fmaf(v.y, W[(k + 1) * 3 + 2], a2);
    a0 = fmaf(v.z, W[(k + 2) * 3 + 0], a0); a1 = fmaf(v.z, W[(k + 2) * 3 + 1], a1); a2 = fmaf(v.z, W[(k + 2) * 3 + 2], a2);
    a0 = fmaf(v.w, W[(k + 3) * 3 + 0], a0); a1 = fmaf(v.w, W[(k + 3) * 3 + 1], a1); a2 = fmaf(v.w, W[(k + 3) * 3 + 2], a2);
  }
  out[n * 3 + 0] = a0; out[n * 3 + 1] = a1; out[n * 3 + 2] = a2;
}

__global__ void fill_zero_kernel(float* __restrict__ p, int n) {
  const int i = blockIdx.x * blockDim.x + threadIdx.x;
  if (i < n) p[i] = 0.0f;
}

// Pack W[K][128] (fp32 row-major) into f16 B-fragments for wmma 16x16x32:
// dword t -> v(3b) | lane(5b) | nt(3b) | kc ; lane = column nt*16+(l&15),
// per-lane K range = kc*32 + (l&16 ? 16 : 0) + [0,16). Zero-pad K>=Kactual.
__global__ void pack_weights_kernel(const float* __restrict__ W, _Float16* __restrict__ dst,
                                    int K, int Kp) {
  const int t = blockIdx.x * blockDim.x + threadIdx.x;
  if (t >= Kp * 64) return;
  const int v  = t & 7;
  const int l  = (t >> 3) & 31;
  const int nt = (t >> 8) & 7;
  const int kc = t >> 11;
  const int k0 = kc * 32 + ((l & 16) ? 16 : 0) + v * 2;
  const int n  = nt * 16 + (l & 15);
  const float x0 = (k0 < K)     ? W[(size_t)k0 * L + n]       : 0.0f;
  const float x1 = (k0 + 1 < K) ? W[(size_t)(k0 + 1) * L + n] : 0.0f;
  union { _Float16 h[2]; uint32_t u; } p;
  p.h[0] = (_Float16)x0; p.h[1] = (_Float16)x1;
  reinterpret_cast<uint32_t*>(dst)[t] = p.u;
}

// ---------------------------------------------------------------------------
extern "C" void kernel_launch(void* const* d_in, const int* in_sizes, int n_in,
                              void* d_out, int out_size, void* d_ws, size_t ws_size,
                              hipStream_t stream) {
  const float* nodes   = (const float*)d_in[0];
  const float* globals = (const float*)d_in[1];
  const int*   senders = (const int*)d_in[2];
  const int*   recvs   = (const int*)d_in[3];
  const float* encW = (const float*)d_in[4];
  const float* encB = (const float*)d_in[5];
  const float* decW = (const float*)d_in[6];
  const float* decB = (const float*)d_in[7];
  const int N = in_sizes[0] / 3;
  const int E = in_sizes[2];

  char* ws = (char*)d_ws;
  size_t cur = 0;
  auto alloc = [&](size_t bytes) -> char* {
    char* p = ws + cur;
    cur = (cur + bytes + 255) & ~(size_t)255;
    return p;
  };
  float*    h      = (float*)alloc((size_t)N * L * 4);
  float*    tmp    = (float*)alloc((size_t)N * L * 4);
  float*    agg    = (float*)alloc((size_t)N * L * 4);
  _Float16* edges  = (_Float16*)alloc((size_t)E * L * 2);

  const int KPe[3] = {288, 416, 416};  // padded edge-MLP K (264, 392, 392)
  const int KPn    = 288;              // padded node-MLP K (264)
  _Float16* eW[3][3];
  _Float16* nW[3][3];
  for (int s = 0; s < 3; ++s) {
    eW[s][0] = (_Float16*)alloc((size_t)KPe[s] * L * 2);
    eW[s][1] = (_Float16*)alloc((size_t)L * L * 2);
    eW[s][2] = (_Float16*)alloc((size_t)L * L * 2);
    nW[s][0] = (_Float16*)alloc((size_t)KPn * L * 2);
    nW[s][1] = (_Float16*)alloc((size_t)L * L * 2);
    nW[s][2] = (_Float16*)alloc((size_t)L * L * 2);
  }

  // --- pack all weights (tiny; re-done every call for determinism) ---
  for (int s = 0; s < 3; ++s) {
    const int pb = 8 + 14 * s;
    const int dinE = (s == 0) ? 264 : 392;
    {
      int tot = KPe[s] * 64;
      pack_weights_kernel<<<(tot + 255) / 256, 256, 0, stream>>>(
          (const float*)d_in[pb + 0], eW[s][0], dinE, KPe[s]);
    }
    for (int l = 1; l < 3; ++l) {
      int tot = L * 64;
      pack_weights_kernel<<<(tot + 255) / 256, 256, 0, stream>>>(
          (const float*)d_in[pb + 2 * l], eW[s][l], L, L);
    }
    {
      int tot = KPn * 64;
      pack_weights_kernel<<<(tot + 255) / 256, 256, 0, stream>>>(
          (const float*)d_in[pb + 6], nW[s][0], 264, KPn);
    }
    for (int l = 1; l < 3; ++l) {
      int tot = L * 64;
      pack_weights_kernel<<<(tot + 255) / 256, 256, 0, stream>>>(
          (const float*)d_in[pb + 6 + 2 * l], nW[s][l], L, L);
    }
  }

  // --- encoder ---
  encode_kernel<<<(N * L + 255) / 256, 256, 0, stream>>>(nodes, encW, encB, h, N);

  const int edgeTiles = (E + 31) / 32;
  const int nodeTiles = (N + 31) / 32;
  int edgeBlocks = (edgeTiles + WAVES - 1) / WAVES; if (edgeBlocks > 2048) edgeBlocks = 2048;
  int nodeBlocks = (nodeTiles + WAVES - 1) / WAVES; if (nodeBlocks > 1024) nodeBlocks = 1024;
  int lnBlocks = (N + 7) / 8; if (lnBlocks > 1024) lnBlocks = 1024;

  for (int s = 0; s < 3; ++s) {
    const int pb = 8 + 14 * s;
    const float* eb1 = (const float*)d_in[pb + 1];
    const float* eb2 = (const float*)d_in[pb + 3];
    const float* eb3 = (const float*)d_in[pb + 5];
    const float* nb1 = (const float*)d_in[pb + 7];
    const float* nb2 = (const float*)d_in[pb + 9];
    const float* nb3 = (const float*)d_in[pb + 11];
    const float* lns = (const float*)d_in[pb + 12];
    const float* lnb = (const float*)d_in[pb + 13];

    fill_zero_kernel<<<(N * L + 255) / 256, 256, 0, stream>>>(agg, N * L);

    if (s == 0) {
      edge_mlp_kernel<288, false><<<edgeBlocks, WAVES * 32, 0, stream>>>(
          h, globals, senders, recvs, edges, agg,
          (const uint4*)eW[s][0], eb1, (const uint4*)eW[s][1], eb2,
          (const uint4*)eW[s][2], eb3, E, edgeTiles);
    } else {
      edge_mlp_kernel<416, true><<<edgeBlocks, WAVES * 32, 0, stream>>>(
          h, globals, senders, recvs, edges, agg,
          (const uint4*)eW[s][0], eb1, (const uint4*)eW[s][1], eb2,
          (const uint4*)eW[s][2], eb3, E, edgeTiles);
    }

    node_mlp_kernel<<<nodeBlocks, WAVES * 32, 0, stream>>>(
        h, agg, globals, tmp,
        (const uint4*)nW[s][0], nb1, (const uint4*)nW[s][1], nb2,
        (const uint4*)nW[s][2], nb3, N, nodeTiles);

    layernorm_kernel<<<lnBlocks, 256, 0, stream>>>(tmp, h, lns, lnb, N);
  }

  // --- decoder ---
  decode_kernel<<<(N + 255) / 256, 256, 0, stream>>>(h, decW, decB, (float*)d_out, N);
}
